// SpectralTransposed2d_31370441130560
// MI455X (gfx1250) — matrix operand order repaired and checked by reading, not compiled, not run on hardware
//
#include <hip/hip_runtime.h>
#include <hip/hip_bf16.h>

// Problem constants
#define BB    8
#define COUT  256
#define CIN   128
#define HH    64
#define WW    64
#define KH    7
#define KW    7
#define WFR   33          // rfft width
#define NTAP  49

typedef __attribute__((ext_vector_type(16))) __bf16 bf16x16;
typedef __attribute__((ext_vector_type(8)))  float  f32x8;

union Frag { uint4 u[2]; bf16x16 v; };

// Workspace layout
#define WT_BYTES ((size_t)NTAP * CIN * COUT * 2)            // 3,211,264 B  [tap][i][o] bf16
#define XT_BYTES ((size_t)BB * HH * WW * COUT * 2)          // 16,777,216 B [b][h][w][o] bf16

// ---------------------------------------------------------------------------
// CDNA5 async LDS helpers (GLOBAL_LOAD_ASYNC_TO_LDS_*, ASYNCcnt; ISA ch.10.7)
// LDS byte offset = low 32 bits of a generic pointer to __shared__ memory
// (flat-addressing rule).
// ---------------------------------------------------------------------------
__device__ __forceinline__ void async_b128(void* lds_dst, const void* gsrc) {
    unsigned l = (unsigned)(unsigned long long)lds_dst;
    unsigned long long g = (unsigned long long)gsrc;
    asm volatile("global_load_async_to_lds_b128 %0, %1, off"
                 :: "v"(l), "v"(g) : "memory");
}
__device__ __forceinline__ void wait_async0() {
    asm volatile("s_wait_asynccnt 0x0" ::: "memory");
}

// ---------------------------------------------------------------------------
// Kernel 1: recover 7x7 spatial weights from weight_fft via inverse rDFT
// evaluated only at the 49 taps. One wave per (o,i) pair; 8 waves / block.
// Writes wT[tap][i][o] in bf16 (the WMMA A-operand friendly layout).
// This pass reads the full 553 MB weight_fft once: it is the HBM bound.
// ---------------------------------------------------------------------------
__global__ void k_recover_weights(const float* __restrict__ wfft,
                                  __bf16* __restrict__ wT) {
    __shared__ float ctab[64];
    const int tid = threadIdx.x;
    if (tid < 64) ctab[tid] = __builtin_cosf(6.28318530717958647692f * (float)tid / 64.0f);
    __syncthreads();

    const int wave = tid >> 5;
    const int lane = tid & 31;
    const int pair = blockIdx.x * 8 + wave;          // 0 .. 32767
    const int o = pair >> 7;                         // 0 .. 255
    const int i = pair & 127;                        // 0 .. 127

    float acc[NTAP];
#pragma unroll
    for (int t = 0; t < NTAP; ++t) acc[t] = 0.0f;

    const float inv = 1.0f / (float)(HH * WW);
    const float* wf = wfft + (((size_t)o * CIN + i) * HH) * (WFR * 2);

    for (int pass = 0; pass < 2; ++pass) {
        const int u = lane + pass * 32;              // frequency row 0..63
        const float* wp = wf + (size_t)u * (WFR * 2);
        for (int v = 0; v < WFR; ++v) {
            float wr = wp[2 * v];
            float wi = wp[2 * v + 1];
            const float sc = (v == 0 || v == WFR - 1) ? inv : 2.0f * inv;
            wr *= sc; wi *= sc;
#pragma unroll
            for (int kh = 0; kh < KH; ++kh) {
                const int tu = (u * kh) & 63;
#pragma unroll
                for (int kw = 0; kw < KW; ++kw) {
                    const int t = (tu + v * kw) & 63;
                    const float c = ctab[t];
                    const float s = ctab[(t + 48) & 63];   // sin = cos(theta - pi/2)
                    acc[kh * KW + kw] += wr * c - wi * s;  // Re(W * e^{i theta})
                }
            }
        }
    }

    // wave32 butterfly reduction
#pragma unroll
    for (int t = 0; t < NTAP; ++t) {
        float v = acc[t];
        for (int off = 16; off > 0; off >>= 1)
            v += __shfl_xor(v, off, 32);
        acc[t] = v;
    }

    if (lane == 0) {
#pragma unroll
        for (int t = 0; t < NTAP; ++t)
            wT[((size_t)t * CIN + i) * COUT + o] = (__bf16)acc[t];
    }
}

// ---------------------------------------------------------------------------
// Kernel 2: transpose x (f32, [b][o][h][w]) -> xT (bf16, [b][h][w][o])
// so that WMMA K-pairs (consecutive o) are contiguous.
// ---------------------------------------------------------------------------
__global__ void k_transpose_x(const float* __restrict__ x,
                              __bf16* __restrict__ xT) {
    __shared__ __align__(16) __bf16 tile[64 * 32];    // [w][o] 4 KB
    const int tid = threadIdx.x;
    const int bx  = blockIdx.x;                       // 8*64*8 = 4096 blocks
    const int oc  = bx & 7;
    const int h   = (bx >> 3) & 63;
    const int b   = bx >> 9;
    const int o0  = oc * 32;

    // read: coalesced over w
#pragma unroll
    for (int r = 0; r < 8; ++r) {
        const int o = r * 4 + (tid >> 6);             // 0..31
        const int w = tid & 63;
        const float v = x[(((size_t)b * COUT + o0 + o) * HH + h) * WW + w];
        tile[w * 32 + o] = (__bf16)v;
    }
    __syncthreads();

    // write: 16B vector stores, o fastest
    const int w = tid >> 2;
    const int q = tid & 3;
    const uint4 d = *(const uint4*)&tile[w * 32 + q * 8];
    __bf16* dst = xT + (((size_t)b * HH + h) * WW + w) * COUT + o0 + q * 8;
    *(uint4*)dst = d;
}

// ---------------------------------------------------------------------------
// Kernel 3: implicit-GEMM circular conv with v_wmma_f32_16x16x32_bf16.
//   M = CIN (tile 32), N = 64 pixels (one image row), K = COUT*49 = 12544.
//   8 waves = 2(M) x 4(N); each wave owns one 16x16 f32 accumulator and
//   issues 7*8*7 = 392 WMMAs. Operands double-buffered in LDS via
//   GLOBAL_LOAD_ASYNC_TO_LDS_B128; 80B row pitch -> 16B-aligned,
//   bank-conflict-free ds_load_b128 fragment gathers.
// ---------------------------------------------------------------------------
#define X_PITCH   80                    // bytes per w row   (32 o * 2B, padded)
#define A_PITCH   80                    // bytes per i row   (32 o * 2B, padded)
#define A_KW      (32 * A_PITCH)        // 2560 B per kw slab
#define XROW_B    (64 * X_PITCH)        // 5120 B
#define A_B       (KW * A_KW)           // 17920 B
#define STAGE_B   (XROW_B + A_B)        // 23040 B
#define NSTAGE    (KH * (COUT / 32))    // 56

__global__ void __launch_bounds__(256)
k_conv_wmma(const __bf16* __restrict__ xT,
            const __bf16* __restrict__ wT,
            const float*  __restrict__ bias,
            float* __restrict__ y) {
    __shared__ __align__(16) unsigned char smem[2][STAGE_B];

    const int tid  = threadIdx.x;
    const int bx   = blockIdx.x;                 // 8*64*4 = 2048 blocks
    const int miB  = bx & 3;                     // which 32-wide i block
    const int h    = (bx >> 2) & 63;
    const int b    = bx >> 8;
    const int mi0  = miB * 32;

    const int wave  = tid >> 5;
    const int lane  = tid & 31;
    const int half  = lane >> 4;
    const int l16   = lane & 15;
    const int mwave = wave >> 2;                 // 0..1 (16 i rows each)
    const int nwave = wave & 3;                  // 0..3 (16 pixels each)
    const int n0    = nwave * 16;
    const int w_pix = n0 + l16;

    // stage-invariant staging coordinates
    const int sw  = tid >> 2;                    // w row for x staging
    const int sq  = tid & 3;                     // 16B chunk within w row
    const int akw = tid >> 5;                    // kw slab for A staging (tid<224)
    const int ail = tid & 31;                    // i row within slab

    auto issue_stage = [&](int kh, int oc, unsigned char* buf) {
        const int hr = (h - kh) & 63;
        const int o0 = oc * 32;
        // x row: 64 w x 32 o (4 KB) -> one async b128 per thread
        const __bf16* xs =
            xT + (((size_t)b * HH + hr) * WW + sw) * COUT + o0 + sq * 8;
        async_b128(buf + sw * X_PITCH + sq * 16, xs);
        // A slab: 7 kw x 32 i x 32 o (14 KB) -> 4 async b128 per thread
        if (tid < KW * 32) {
            const int tap = kh * KW + akw;
            const __bf16* as =
                wT + ((size_t)tap * CIN + mi0 + ail) * COUT + o0;
            unsigned char* ad = buf + XROW_B + akw * A_KW + ail * A_PITCH;
#pragma unroll
            for (int q = 0; q < 4; ++q)
                async_b128(ad + q * 16, as + q * 8);
        }
    };

    f32x8 acc = {};

    // prologue: fill buffer 0 with stage 0
    issue_stage(0, 0, smem[0]);

    int p = 0;
    for (int s = 0; s < NSTAGE; ++s) {
        wait_async0();          // my stage-s async loads have landed in LDS
        __syncthreads();        // everyone's have; everyone done reading buf p^1

        if (s + 1 < NSTAGE) {   // prefetch next stage into the other buffer
            const int sn = s + 1;
            issue_stage(sn >> 3, sn & 7, smem[p ^ 1]);
        }

        unsigned char* ldsX = smem[p];
        unsigned char* ldsA = smem[p] + XROW_B;

#pragma unroll
        for (int kw = 0; kw < KW; ++kw) {
            Frag A, B;
            // A fragment: lane = M row; K groups at half*16 and half*16+32
            const unsigned char* ap =
                ldsA + kw * A_KW + (mwave * 16 + l16) * A_PITCH + half * 16;
            A.u[0] = *(const uint4*)(ap);
            A.u[1] = *(const uint4*)(ap + 32);
            // B fragment: lane = N col; K groups at half*16 and half*16+32 bytes
            const int w_src = (w_pix - kw) & 63;
            const unsigned char* bp = ldsX + w_src * X_PITCH + half * 16;
            B.u[0] = *(const uint4*)(bp);
            B.u[1] = *(const uint4*)(bp + 32);

            acc = __builtin_amdgcn_wmma_f32_16x16x32_bf16(
                      false, A.v, false, B.v, (short)0, acc, false, false);
        }
        p ^= 1;
    }

    // epilogue: C/D layout -> lane n = l16, VGPR v -> M = v + half*8
#pragma unroll
    for (int v = 0; v < 8; ++v) {
        const int ig = mi0 + mwave * 16 + half * 8 + v;
        y[(((size_t)b * CIN + ig) * HH + h) * WW + w_pix] = acc[v] + bias[ig];
    }
}

// ---------------------------------------------------------------------------
extern "C" void kernel_launch(void* const* d_in, const int* in_sizes, int n_in,
                              void* d_out, int out_size, void* d_ws, size_t ws_size,
                              hipStream_t stream) {
    const float* x    = (const float*)d_in[0];
    const float* wfft = (const float*)d_in[1];   // complex64 interleaved
    const float* bias = (const float*)d_in[2];
    float* y = (float*)d_out;

    __bf16* wT = (__bf16*)d_ws;
    __bf16* xT = (__bf16*)((char*)d_ws + WT_BYTES);

    // 32768 (o,i) pairs, 8 waves/block
    k_recover_weights<<<4096, 256, 0, stream>>>(wfft, wT);
    // 8 b * 64 h * 8 o-chunks
    k_transpose_x<<<4096, 256, 0, stream>>>(x, xT);
    // 8 b * 64 h * 4 i-blocks
    k_conv_wmma<<<2048, 256, 0, stream>>>(xT, wT, bias, y);
}
